// HMSS_72980084293806
// MI455X (gfx1250) — matrix operand order repaired and checked by reading, not compiled, not run on hardware
//
#include <hip/hip_runtime.h>
#include <hip/hip_bf16.h>
#include <math.h>

#define B_   8
#define L_   512
#define D_   256
#define NL_  2
#define NS_  16
#define DC_  4
#define DI_  512
#define DTR_ 16
#define T_   (B_ * L_)

static constexpr float EPS = 1e-5f;

typedef __attribute__((ext_vector_type(16))) __bf16 v16bf;
typedef __attribute__((ext_vector_type(8)))  float  v8f;

__device__ __forceinline__ float sigmoidf_(float x) { return 1.f / (1.f + __expf(-x)); }
__device__ __forceinline__ float softplusf_(float x) { return x > 20.f ? x : log1pf(__expf(x)); }

// ---------------------------------------------------------------------------
// WMMA fragment layouts (CDNA5 wave32, 16-bit data, 16x16x32):
// A (16xK): lane<16 -> row=lane,   elems 0..15 = K+0..15 (contiguous k order, as loaded here)
//           lane>=16 -> row=lane-16, elems 0..15 = K+16..31
// B (Kx16 from W[N,K]): lane half selects K 0..15 / 16..31 of column N.
// ---------------------------------------------------------------------------
__device__ __forceinline__ v16bf frag_from(const __bf16* lo_p, const __bf16* hi_p) {
  union { uint4 u; __bf16 h[8]; } lo, hi;
  lo.u = *(const uint4*)(lo_p);
  hi.u = *(const uint4*)(hi_p);
  v16bf r;
#pragma unroll
  for (int j = 0; j < 8; ++j) { r[j] = lo.h[j]; r[8 + j] = hi.h[j]; }
  return r;
}

// ---------------------------------------------------------------------------
// LDS-staged GEMM: C[M,N] = act(A[M,K] @ W[N,K]^T + bias) + resid
// Block = 8 waves -> 64x64 macro-tile (wave = 1 m-tile x 2 n-tiles, A-frag
// reused). K pipelined 32-wide, double-buffered LDS filled with
// ASYNCcnt-tracked global_load_async_to_lds_b128 DMA; K templated for full
// unroll/software pipelining. Requires: M%64==0, N%64==0.
// ---------------------------------------------------------------------------
template <int K>
__global__ __launch_bounds__(256) void gemm_bf16_wmma_lds(
    const __bf16* __restrict__ A, const __bf16* __restrict__ W,
    const float* __restrict__ bias, const float* __restrict__ resid,
    float* __restrict__ C, int M, int Nn, int act)
{
  __shared__ __bf16 sA[2][64 * 32];   // 8 KB
  __shared__ __bf16 sW[2][64 * 32];   // 8 KB

  const int tid  = threadIdx.x;
  const int wrp  = tid >> 5;
  const int lane = tid & 31;
  const int half = (lane >> 4) & 1;
  const int l15  = lane & 15;
  const int m_blk = blockIdx.y * 64;
  const int n_blk = blockIdx.x * 64;
  const int mt = wrp & 3;             // 0..3  m sub-tile
  const int nh = wrp >> 2;            // 0..1  n half (2 tiles each)

  // LDS byte offsets for async-DMA VDST (flat shared addr truncates to LDS offset)
  const unsigned sA0 = (unsigned)(uintptr_t)(&sA[0][0]);
  const unsigned sW0 = (unsigned)(uintptr_t)(&sW[0][0]);

  const __bf16* gA = A + (size_t)m_blk * K;
  const __bf16* gW = W + (size_t)n_blk * K;
  const int prow = tid >> 2;          // 0..63
  const int pseg = tid & 3;           // 16B segment within a 64B panel row

  auto issue = [&](int kk, int buf) {
    // A panel: 64 rows x 32 bf16 = 256 x 16B; W panel same shape
    const __bf16* gpa = gA + (size_t)prow * K + kk + pseg * 8;
    unsigned la = sA0 + (unsigned)buf * (64 * 32 * 2) + prow * 64 + pseg * 16;
    asm volatile("global_load_async_to_lds_b128 %0, %1, off"
                 :: "v"(la), "v"(gpa) : "memory");
    const __bf16* gpw = gW + (size_t)prow * K + kk + pseg * 8;
    unsigned lw = sW0 + (unsigned)buf * (64 * 32 * 2) + prow * 64 + pseg * 16;
    asm volatile("global_load_async_to_lds_b128 %0, %1, off"
                 :: "v"(lw), "v"(gpw) : "memory");
  };

  issue(0, 0);
  v8f acc0 = {}, acc1 = {};
#pragma unroll
  for (int kk = 0; kk < K; kk += 32) {
    const int buf = (kk >> 5) & 1;
    asm volatile("s_wait_asynccnt 0x0" ::: "memory");  // own DMA landed in LDS
    __syncthreads();                                    // all waves' panels visible
    if (kk + 32 < K) issue(kk + 32, buf ^ 1);           // prefetch next K-slice

    const __bf16* la  = &sA[buf][(mt * 16 + l15) * 32 + half * 16];
    const __bf16* lw0 = &sW[buf][(nh * 32 + l15) * 32 + half * 16];
    const __bf16* lw1 = &sW[buf][(nh * 32 + 16 + l15) * 32 + half * 16];
    v16bf af  = frag_from(la, la + 8);     // ds_load_b128 pair, reused twice
    v16bf wf0 = frag_from(lw0, lw0 + 8);
    v16bf wf1 = frag_from(lw1, lw1 + 8);
    acc0 = __builtin_amdgcn_wmma_f32_16x16x32_bf16(
        false, af, false, wf0, (short)0, acc0, false, false);
    acc1 = __builtin_amdgcn_wmma_f32_16x16x32_bf16(
        false, af, false, wf1, (short)0, acc1, false, false);
    __syncthreads();                                    // reads done before rewrite
  }

  const int n0    = n_blk + nh * 32 + l15;
  const int n1    = n0 + 16;
  const int mbase = m_blk + mt * 16 + half * 8;
  const float bv0 = bias ? bias[n0] : 0.f;
  const float bv1 = bias ? bias[n1] : 0.f;
#pragma unroll
  for (int i = 0; i < 8; ++i) {
    const int m = mbase + i;
    float v0 = acc0[i] + bv0;
    float v1 = acc1[i] + bv1;
    if (act == 1) { v0 = softplusf_(v0); v1 = softplusf_(v1); }
    if (resid) {
      v0 += resid[(size_t)m * Nn + n0];
      v1 += resid[(size_t)m * Nn + n1];
    }
    C[(size_t)m * Nn + n0] = v0;
    C[(size_t)m * Nn + n1] = v1;
  }
}

// ---------------------------------------------------------------------------
// Simple per-wave GEMM (for N=48 xproj, N not a multiple of 64)
// ---------------------------------------------------------------------------
__global__ __launch_bounds__(256) void gemm_bf16_wmma(
    const __bf16* __restrict__ A, const __bf16* __restrict__ W,
    const float* __restrict__ bias, const float* __restrict__ resid,
    float* __restrict__ C, int M, int Nn, int K, int act)
{
  const int wave = (blockIdx.x * blockDim.x + threadIdx.x) >> 5;
  const int lane = threadIdx.x & 31;
  const int tiles_n = Nn >> 4;
  const int tiles_m = M >> 4;
  const int tm = wave / tiles_n;
  const int tn = wave % tiles_n;
  if (tm >= tiles_m) return;

  const int half = (lane >> 4) & 1;
  const int l15  = lane & 15;
  const __bf16* Ap = A + (size_t)(tm * 16 + l15) * K + half * 16;
  const __bf16* Wp = W + (size_t)(tn * 16 + l15) * K + half * 16;

  v8f acc = {};
  for (int k = 0; k < K; k += 32) {
    __builtin_prefetch(Ap + k + 64, 0, 1);
    __builtin_prefetch(Wp + k + 64, 0, 1);
    v16bf af = frag_from(Ap + k, Ap + k + 8);
    v16bf wf = frag_from(Wp + k, Wp + k + 8);
    acc = __builtin_amdgcn_wmma_f32_16x16x32_bf16(
        false, af, false, wf, (short)0, acc, false, false);
  }

  const int n     = tn * 16 + l15;
  const int mbase = tm * 16 + half * 8;
  const float bv  = bias ? bias[n] : 0.f;
#pragma unroll
  for (int i = 0; i < 8; ++i) {
    const int m = mbase + i;
    float v = acc[i] + bv;
    if (act == 1) v = softplusf_(v);
    if (resid) v += resid[(size_t)m * Nn + n];
    C[(size_t)m * Nn + n] = v;
  }
}

// ---------------------------------------------------------------------------
// Elementwise / fused kernels
// ---------------------------------------------------------------------------
__global__ void convert_f32_bf16(const float* __restrict__ s, __bf16* __restrict__ d, int n) {
  int i = blockIdx.x * blockDim.x + threadIdx.x;
  if (i < n) d[i] = (__bf16)s[i];
}

__global__ __launch_bounds__(256) void layernorm_bf16_kernel(
    const float* __restrict__ x, const float* __restrict__ g,
    const float* __restrict__ b, const float* __restrict__ gate,
    __bf16* __restrict__ out)
{
  __shared__ float part[8];
  const int t = blockIdx.x, c = threadIdx.x;
  float v = x[(size_t)t * D_ + c];
  float s = v;
#pragma unroll
  for (int o = 16; o; o >>= 1) s += __shfl_xor(s, o, 32);
  if ((c & 31) == 0) part[c >> 5] = s;
  __syncthreads();
  float tot = 0.f;
#pragma unroll
  for (int i = 0; i < 8; ++i) tot += part[i];
  const float mean = tot * (1.f / 256.f);
  __syncthreads();
  const float d0 = v - mean;
  float q = d0 * d0;
#pragma unroll
  for (int o = 16; o; o >>= 1) q += __shfl_xor(q, o, 32);
  if ((c & 31) == 0) part[c >> 5] = q;
  __syncthreads();
  float var = 0.f;
#pragma unroll
  for (int i = 0; i < 8; ++i) var += part[i];
  var *= (1.f / 256.f);
  float y = d0 * rsqrtf(var + EPS) * g[c] + b[c];
  if (gate) y *= gate[(size_t)t * D_ + c];
  out[(size_t)t * D_ + c] = (__bf16)y;
}

__global__ __launch_bounds__(256) void rmsnorm_bf16_kernel(
    const float* __restrict__ x, const float* __restrict__ w, __bf16* __restrict__ out)
{
  __shared__ float part[8];
  const int t = blockIdx.x, c = threadIdx.x;
  float v = x[(size_t)t * D_ + c];
  float s = v * v;
#pragma unroll
  for (int o = 16; o; o >>= 1) s += __shfl_xor(s, o, 32);
  if ((c & 31) == 0) part[c >> 5] = s;
  __syncthreads();
  float tot = 0.f;
#pragma unroll
  for (int i = 0; i < 8; ++i) tot += part[i];
  out[(size_t)t * D_ + c] = (__bf16)(v * rsqrtf(tot * (1.f / 256.f) + EPS) * w[c]);
}

// im2col for the 3-tap channel-mixing conv (padding 1,1)
__global__ void im2col_kernel(const float* __restrict__ x1, __bf16* __restrict__ col) {
  int idx = blockIdx.x * blockDim.x + threadIdx.x;
  if (idx >= T_ * 768) return;
  const int kc = idx % 768, tt = idx / 768;
  const int b = tt / L_, t = tt % L_;
  const int k = kc / D_, c = kc % D_;
  const int ts = t + k - 1;
  float v = 0.f;
  if (ts >= 0 && ts < L_) v = x1[((size_t)(b * L_ + ts)) * D_ + c];
  col[idx] = (__bf16)v;
}

// hconv_w (O,I,3) -> row-major (O, 768) with [k*256+i]
__global__ void hconv_repack_kernel(const float* __restrict__ w, __bf16* __restrict__ out) {
  int idx = blockIdx.x * blockDim.x + threadIdx.x;
  if (idx >= D_ * 768) return;
  const int o = idx / 768, r = idx % 768;
  const int k = r / D_, i = r % D_;
  out[idx] = (__bf16)w[((size_t)o * D_ + i) * 3 + k];
}

// dtw (NL*DI,16) -> zero-padded (NL*DI,32) bf16
__global__ void pad_dtw_kernel(const float* __restrict__ w, __bf16* __restrict__ out) {
  int idx = blockIdx.x * blockDim.x + threadIdx.x;
  if (idx >= NL_ * DI_ * 32) return;
  const int j = idx & 31, r = idx >> 5;
  out[idx] = (j < DTR_) ? (__bf16)w[r * DTR_ + j] : (__bf16)0.f;
}

// dbc[:, :16] -> zero-padded (T,32) bf16 GEMM input
__global__ void dt_in_kernel(const float* __restrict__ dbc, __bf16* __restrict__ out) {
  int idx = blockIdx.x * blockDim.x + threadIdx.x;
  if (idx >= T_ * 32) return;
  const int j = idx & 31, tt = idx >> 5;
  out[idx] = (j < DTR_) ? (__bf16)dbc[(size_t)tt * 48 + j] : (__bf16)0.f;
}

// Causal depthwise conv (DC=4) over xi half of xz + SiLU; emits f32 + bf16.
__global__ void dwconv_silu_kernel(
    const float* __restrict__ xz, const float* __restrict__ cw,
    const float* __restrict__ cb, float* __restrict__ xi_f, __bf16* __restrict__ xi_bf)
{
  int idx = blockIdx.x * blockDim.x + threadIdx.x;
  if (idx >= T_ * DI_) return;
  const int d = idx % DI_, tt = idx / DI_;
  const int b = tt / L_, t = tt % L_;
  float acc = cb[d];
#pragma unroll
  for (int k = 0; k < DC_; ++k) {
    const int ts = t + k - (DC_ - 1);
    if (ts >= 0) acc += xz[((size_t)(b * L_ + ts)) * (2 * DI_) + d] * cw[d * DC_ + k];
  }
  const float s = acc * sigmoidf_(acc);
  xi_f[idx] = s;
  xi_bf[idx] = (__bf16)s;
}

// Selective-scan recurrence + skip + SiLU(z) gating, fused; bf16 output.
__global__ __launch_bounds__(256) void scan_gate_kernel(
    const float* __restrict__ delta, const float* __restrict__ xi,
    const float* __restrict__ dbc, const float* __restrict__ Alog,
    const float* __restrict__ Dp, const float* __restrict__ xz,
    __bf16* __restrict__ outg)
{
  const int d = blockIdx.x * blockDim.x + threadIdx.x;
  const int b = blockIdx.y;
  if (d >= DI_) return;
  float A[NS_], h[NS_];
#pragma unroll
  for (int n = 0; n < NS_; ++n) { A[n] = -__expf(Alog[(size_t)d * NS_ + n]); h[n] = 0.f; }
  const float Dd = Dp[d];
  for (int t = 0; t < L_; ++t) {
    const size_t idx = (size_t)(b * L_ + t);
    const float dt = delta[idx * DI_ + d];
    const float xv = xi[idx * DI_ + d];
    const float* bc = dbc + idx * 48;
    const float dx = dt * xv;
    float y = 0.f;
#pragma unroll
    for (int n = 0; n < NS_; ++n) {
      h[n] = __expf(dt * A[n]) * h[n] + dx * bc[16 + n];
      y = fmaf(h[n], bc[32 + n], y);
    }
    y = fmaf(Dd, xv, y);
    const float zv = xz[idx * (2 * DI_) + DI_ + d];
    outg[idx * DI_ + d] = (__bf16)(y * zv * sigmoidf_(zv));
  }
}

__global__ void flip_kernel(const float* __restrict__ src, float* __restrict__ dst) {
  int idx = blockIdx.x * blockDim.x + threadIdx.x;
  if (idx >= T_ * D_) return;
  const int c = idx % D_, tt = idx / D_;
  const int b = tt / L_, t = tt % L_;
  dst[idx] = src[((size_t)(b * L_ + (L_ - 1 - t))) * D_ + c];
}

// cat[t,0:256]=xf; cat[t,256:512]=flip(xb_rev) (un-flip fused into the read)
__global__ void concat_bf_kernel(const float* __restrict__ xf, const float* __restrict__ xb_rev,
                                 __bf16* __restrict__ cat) {
  int idx = blockIdx.x * blockDim.x + threadIdx.x;
  if (idx >= T_ * 2 * D_) return;
  const int c = idx % (2 * D_), tt = idx / (2 * D_);
  float v;
  if (c < D_) v = xf[(size_t)tt * D_ + c];
  else {
    const int b = tt / L_, t = tt % L_;
    v = xb_rev[((size_t)(b * L_ + (L_ - 1 - t))) * D_ + (c - D_)];
  }
  cat[idx] = (__bf16)v;
}

// ---------------------------------------------------------------------------
// Host orchestration
// ---------------------------------------------------------------------------
extern "C" void kernel_launch(void* const* d_in, const int* in_sizes, int n_in,
                              void* d_out, int out_size, void* d_ws, size_t ws_size,
                              hipStream_t stream)
{
  (void)in_sizes; (void)n_in; (void)out_size; (void)ws_size;

  const float* x       = (const float*)d_in[0];
  const float* ln_g    = (const float*)d_in[1];
  const float* ln_b    = (const float*)d_in[2];
  const float* w1      = (const float*)d_in[3];
  const float* b1      = (const float*)d_in[4];
  const float* w2      = (const float*)d_in[5];
  const float* b2      = (const float*)d_in[6];
  const float* w3      = (const float*)d_in[7];
  const float* b3      = (const float*)d_in[8];
  const float* down_w  = (const float*)d_in[9];
  const float* down_b  = (const float*)d_in[10];
  const float* hconv_w = (const float*)d_in[11];
  const float* hconv_b = (const float*)d_in[12];
  // mp[dir][j]: 0 inproj, 1 convw, 2 convb, 3 xproj, 4 dtw, 5 dtb, 6 Alog, 7 D, 8 outw, 9 rms
  const float* mp[2][10];
  for (int dir = 0; dir < 2; ++dir)
    for (int j = 0; j < 10; ++j) mp[dir][j] = (const float*)d_in[13 + dir * 10 + j];

  char* wp = (char*)d_ws;
  auto alloc = [&](size_t bytes) -> void* {
    void* r = (void*)wp;
    wp += (bytes + 255) & ~(size_t)255;
    return r;
  };

  float*  x1    = (float*)alloc((size_t)T_ * D_ * 4);
  float*  x2    = (float*)alloc((size_t)T_ * D_ * 4);
  float*  xc    = (float*)alloc((size_t)T_ * D_ * 4);
  float*  xfb   = (float*)alloc((size_t)T_ * D_ * 4);
  float*  xbb   = (float*)alloc((size_t)T_ * D_ * 4);
  float*  dbuf  = (float*)alloc((size_t)T_ * D_ * 4);
  float*  xz    = (float*)alloc((size_t)T_ * 2 * DI_ * 4);
  float*  xi_f  = (float*)alloc((size_t)T_ * DI_ * 4);
  float*  dlt   = (float*)alloc((size_t)T_ * DI_ * 4);
  float*  dbc   = (float*)alloc((size_t)T_ * 48 * 4);
  __bf16* xlbf  = (__bf16*)alloc((size_t)T_ * D_ * 2);
  __bf16* abf   = (__bf16*)alloc((size_t)T_ * DI_ * 2);
  __bf16* colbf = (__bf16*)alloc((size_t)T_ * 768 * 2);
  __bf16* dtinb = (__bf16*)alloc((size_t)T_ * 32 * 2);
  __bf16* w1b   = (__bf16*)alloc((size_t)D_ * D_ * 2);
  __bf16* w2b   = (__bf16*)alloc((size_t)D_ * D_ * 2);
  __bf16* w3b   = (__bf16*)alloc((size_t)D_ * D_ * 2);
  __bf16* dwnb  = (__bf16*)alloc((size_t)D_ * 2 * D_ * 2);
  __bf16* hwb   = (__bf16*)alloc((size_t)D_ * 768 * 2);
  __bf16* inpb[2], *xprjb[2], *dtwb[2], *outwb[2];
  for (int dir = 0; dir < 2; ++dir) {
    inpb[dir]  = (__bf16*)alloc((size_t)NL_ * 2 * DI_ * D_ * 2);
    xprjb[dir] = (__bf16*)alloc((size_t)NL_ * 48 * DI_ * 2);
    dtwb[dir]  = (__bf16*)alloc((size_t)NL_ * DI_ * 32 * 2);
    outwb[dir] = (__bf16*)alloc((size_t)NL_ * D_ * DI_ * 2);
  }

  auto cvt = [&](const float* s, __bf16* d, int n) {
    convert_f32_bf16<<<(n + 255) / 256, 256, 0, stream>>>(s, d, n);
  };
  auto gemm = [&](const __bf16* A, const __bf16* W, const float* bias,
                  const float* resid, float* C, int M, int Nn, int K, int act) {
    if ((Nn % 64) == 0 && (M % 64) == 0 &&
        (K == 32 || K == 256 || K == 512 || K == 768)) {
      dim3 g(Nn / 64, M / 64);
      switch (K) {
        case 32:
          gemm_bf16_wmma_lds<32><<<g, 256, 0, stream>>>(A, W, bias, resid, C, M, Nn, act);
          break;
        case 256:
          gemm_bf16_wmma_lds<256><<<g, 256, 0, stream>>>(A, W, bias, resid, C, M, Nn, act);
          break;
        case 512:
          gemm_bf16_wmma_lds<512><<<g, 256, 0, stream>>>(A, W, bias, resid, C, M, Nn, act);
          break;
        default:
          gemm_bf16_wmma_lds<768><<<g, 256, 0, stream>>>(A, W, bias, resid, C, M, Nn, act);
          break;
      }
    } else {
      const int tiles = (M / 16) * (Nn / 16);
      gemm_bf16_wmma<<<(tiles + 7) / 8, 256, 0, stream>>>(A, W, bias, resid, C, M, Nn, K, act);
    }
  };

  // --- weight conversion (bf16) ---
  cvt(w1, w1b, D_ * D_);
  cvt(w2, w2b, D_ * D_);
  cvt(w3, w3b, D_ * D_);
  cvt(down_w, dwnb, D_ * 2 * D_);
  hconv_repack_kernel<<<(D_ * 768 + 255) / 256, 256, 0, stream>>>(hconv_w, hwb);
  for (int dir = 0; dir < 2; ++dir) {
    cvt(mp[dir][0], inpb[dir], NL_ * 2 * DI_ * D_);
    cvt(mp[dir][3], xprjb[dir], NL_ * 48 * DI_);
    cvt(mp[dir][8], outwb[dir], NL_ * D_ * DI_);
    pad_dtw_kernel<<<(NL_ * DI_ * 32 + 255) / 256, 256, 0, stream>>>(mp[dir][4], dtwb[dir]);
  }

  // --- front: LN, w1/w2, channel-mixing conv as K=768 GEMM ---
  layernorm_bf16_kernel<<<T_, 256, 0, stream>>>(x, ln_g, ln_b, nullptr, xlbf);
  gemm(xlbf, w1b, b1, nullptr, x1, T_, D_, D_, 0);
  gemm(xlbf, w2b, b2, nullptr, x2, T_, D_, D_, 0);
  im2col_kernel<<<(T_ * 768 + 255) / 256, 256, 0, stream>>>(x1, colbf);
  gemm(colbf, hwb, hconv_b, nullptr, xc, T_, D_, 768, 0);

  // --- mamba stacks (dir 0 forward on xc; dir 1 on flipped xc) ---
  auto run_mamba = [&](int dir, float* xcur) {
    for (int l = 0; l < NL_; ++l) {
      rmsnorm_bf16_kernel<<<T_, 256, 0, stream>>>(xcur, mp[dir][9] + l * D_, xlbf);
      gemm(xlbf, inpb[dir] + (size_t)l * 2 * DI_ * D_, nullptr, nullptr, xz, T_, 2 * DI_, D_, 0);
      dwconv_silu_kernel<<<(T_ * DI_ + 255) / 256, 256, 0, stream>>>(
          xz, mp[dir][1] + l * DI_ * DC_, mp[dir][2] + l * DI_, xi_f, abf);
      gemm(abf, xprjb[dir] + (size_t)l * 48 * DI_, nullptr, nullptr, dbc, T_, 48, DI_, 0);
      dt_in_kernel<<<(T_ * 32 + 255) / 256, 256, 0, stream>>>(dbc, dtinb);
      gemm(dtinb, dtwb[dir] + (size_t)l * DI_ * 32, mp[dir][5] + l * DI_, nullptr,
           dlt, T_, DI_, 32, 1);
      scan_gate_kernel<<<dim3(DI_ / 256, B_), 256, 0, stream>>>(
          dlt, xi_f, dbc, mp[dir][6] + (size_t)l * DI_ * NS_, mp[dir][7] + l * DI_, xz, abf);
      gemm(abf, outwb[dir] + (size_t)l * D_ * DI_, nullptr, xcur, xcur, T_, D_, DI_, 0);
    }
  };

  hipMemcpyAsync(xfb, xc, (size_t)T_ * D_ * 4, hipMemcpyDeviceToDevice, stream);
  run_mamba(0, xfb);
  flip_kernel<<<(T_ * D_ + 255) / 256, 256, 0, stream>>>(xc, xbb);
  run_mamba(1, xbb);

  // --- tail: concat (un-flip fused), down, LN*x2 gate, w3 + residual ---
  concat_bf_kernel<<<(T_ * 2 * D_ + 255) / 256, 256, 0, stream>>>(xfb, xbb, abf);
  gemm(abf, dwnb, down_b, nullptr, dbuf, T_, D_, 2 * D_, 0);
  layernorm_bf16_kernel<<<T_, 256, 0, stream>>>(dbuf, ln_g, ln_b, x2, xlbf);
  gemm(xlbf, w3b, b3, x, (float*)d_out, T_, D_, D_, 0);
}
